// DMPNN_47510928228864
// MI455X (gfx1250) — matrix operand order repaired and checked by reading, not compiled, not run on hardware
//
#include <hip/hip_runtime.h>
#include <hip/hip_bf16.h>
#include <stdint.h>

// ---------------- problem constants (match reference) ----------------
#define N_ATOMS   200000
#define N_BONDS   400000
#define MAX_NB    4
#define N_MOLS    10000
#define ATOM_FDIM 133
#define BOND_FDIM 147          // BOND_FDIM_FULL
#define HIDDEN    128
#define DEPTH     6
#define FFN_HIDDEN 256

typedef float  v8f   __attribute__((ext_vector_type(8)));
typedef float  v2f   __attribute__((ext_vector_type(2)));
typedef __bf16 v16bf __attribute__((ext_vector_type(16)));
typedef __bf16 v2bf  __attribute__((ext_vector_type(2)));

union Frag16 { unsigned u[8]; v16bf v; };
union Q4     { uint4 q; unsigned u[4]; };
union PK2    { v2bf h; unsigned u; unsigned short s[2]; };

// packed f32x2 -> bf16x2 (lowers to v_cvt_pk_bf16_f32, RNE)
__device__ __forceinline__ unsigned pack_bf16x2(float a, float b) {
  PK2 p;
  p.h = __builtin_convertvector((v2f){a, b}, v2bf);
  return p.u;
}
__device__ __forceinline__ unsigned short f2bf(float x) {
  PK2 p;
  p.h = __builtin_convertvector((v2f){x, 0.0f}, v2bf);
  return p.s[0];
}
__device__ __forceinline__ float bflo(unsigned p) { return __uint_as_float(p << 16); }
__device__ __forceinline__ float bfhi(unsigned p) { return __uint_as_float(p & 0xFFFF0000u); }
// packed bf16 pair ops in f32, repacked with HW cvt (feeds WMMA at bf16 anyway)
__device__ __forceinline__ unsigned subpk(unsigned a, unsigned b) {
  return pack_bf16x2(bflo(a) - bflo(b), bfhi(a) - bfhi(b));
}
__device__ __forceinline__ unsigned addpk4(unsigned a, unsigned b, unsigned c, unsigned d) {
  return pack_bf16x2(bflo(a) + bflo(b) + bflo(c) + bflo(d),
                     bfhi(a) + bfhi(b) + bfhi(c) + bfhi(d));
}

// =====================================================================
// Fused WMMA GEMM:  C[M,128] = epilogue( A(M,K) @ W(K,128) )
//  AMODE 0: fp32 rows A0+m*lda, K-guarded tail  (f_bonds K=147, f_atoms K=133)
//  AMODE 1: bf16 gather-sub A0[idx0[m]] - A1[idx1[m]]   (message passing)
//  AMODE 2: fp32 rows scaled by 1/count (segment mean head)
//  AMODE 3: bf16 plain rows (a_message @ W_o bottom half)
//  Epilogue: v = acc (+addend) (+bias); C2 = pre-act; C = relu?(v)
//  CBF16: bf16 vs fp32 C/C2/addend.  MFULL: M is a multiple of 128.
// =====================================================================
template<int AMODE, int KPAD, int KACT, bool CBF16, bool MFULL>
__launch_bounds__(256)
__global__ void gemm_wmma_bf16(
    const void* __restrict__ A0v, const void* __restrict__ A1v,
    const int* __restrict__ idx0, const int* __restrict__ idx1,
    const float* __restrict__ W, const float* __restrict__ bias,
    const void* addendv, void* Cv, void* C2v,
    int M, int lda, int do_relu, int ldw, int ldc)
{
  static_assert(KPAD % 32 == 0, "KPAD multiple of 32");
  constexpr int N = 128, NT = 8, KP2 = KPAD / 2, NC = KPAD / 32;
  __shared__ unsigned ldsw[N * KP2];

  const int tid = threadIdx.x;
  // ---- stage W -> LDS as packed bf16 K-pairs (ISA B-matrix layout) ----
  for (int i = tid; i < N * KP2; i += 256) {
    const int n  = i / KP2;
    const int k0 = (i - n * KP2) * 2;
    const float w0 = (k0     < KACT) ? W[(size_t)k0       * ldw + n] : 0.0f;
    const float w1 = (k0 + 1 < KACT) ? W[(size_t)(k0 + 1) * ldw + n] : 0.0f;
    ldsw[i] = pack_bf16x2(w0, w1);
  }
  __syncthreads();

  const int wave  = tid >> 5;
  const int lane  = tid & 31;
  const int khalf = lane >> 4;
  const int nlo   = lane & 15;
  const int m0    = blockIdx.x * 128 + wave * 16;
  int mlane = m0 + nlo;
  if (!MFULL && mlane > M - 1) mlane = M - 1;  // clamp: EXEC stays all-ones

  const float*          rowaf = nullptr;
  const unsigned short* rowah = nullptr;
  const unsigned short* rowbh = nullptr;
  float scale = 1.0f;
  if constexpr (AMODE == 0) {
    rowaf = (const float*)A0v + (size_t)mlane * lda;
  } else if constexpr (AMODE == 1) {
    rowah = (const unsigned short*)A0v + (size_t)idx0[mlane] * HIDDEN;
    rowbh = (const unsigned short*)A1v + (size_t)idx1[mlane] * HIDDEN;
  } else if constexpr (AMODE == 2) {
    rowaf = (const float*)A0v + (size_t)mlane * HIDDEN;
    const float c = ((const float*)A1v)[mlane];
    scale = (c > 0.0f) ? (1.0f / c) : 0.0f;
  } else {
    rowah = (const unsigned short*)A0v + (size_t)mlane * HIDDEN;
  }

  // prefetch the epilogue addend tile while the K loop runs
  // (32 lanes x 128B cachelines cover the 16x128 bf16 tile)
  if (addendv) {
    if constexpr (CBF16)
      __builtin_prefetch((const unsigned short*)addendv + (size_t)mlane * ldc + khalf * 64, 0, 1);
    else
      __builtin_prefetch((const float*)addendv + (size_t)mlane * ldc + khalf * 64, 0, 1);
  }

  v8f acc[NT] = {};

  #pragma unroll
  for (int kc = 0; kc < NC; ++kc) {
    const int kb = kc * 32 + khalf * 8;   // A layout: K = kb..kb+7, kb+16..kb+23
    Frag16 a;
    if constexpr (AMODE == 0 || AMODE == 2) {
      float f[16];
      if (kc * 32 + 31 < KACT) {          // constant-folded after full unroll
        #pragma unroll
        for (int e = 0; e < 16; ++e) f[e] = rowaf[kb + (e < 8 ? e : e + 8)];
      } else {                            // tail: branchless clamp + select
        #pragma unroll
        for (int e = 0; e < 16; ++e) {
          const int kk = kb + (e < 8 ? e : e + 8);
          const int ks = (kk < KACT) ? kk : 0;
          const float v = rowaf[ks];
          f[e] = (kk < KACT) ? v : 0.0f;
        }
      }
      if constexpr (AMODE == 2) {
        #pragma unroll
        for (int e = 0; e < 16; ++e) f[e] *= scale;
      }
      #pragma unroll
      for (int j = 0; j < 8; ++j) a.u[j] = pack_bf16x2(f[2 * j], f[2 * j + 1]);
    } else if constexpr (AMODE == 1) {
      Q4 xa, xb, ya, yb;
      xa.q = *(const uint4*)(rowah + kb);
      xb.q = *(const uint4*)(rowah + kb + 16);
      ya.q = *(const uint4*)(rowbh + kb);
      yb.q = *(const uint4*)(rowbh + kb + 16);
      #pragma unroll
      for (int j = 0; j < 4; ++j) { a.u[j] = subpk(xa.u[j], ya.u[j]);
                                    a.u[4 + j] = subpk(xb.u[j], yb.u[j]); }
    } else {  // AMODE 3: raw bf16 rows, zero conversion cost
      Q4 xa, xb;
      xa.q = *(const uint4*)(rowah + kb);
      xb.q = *(const uint4*)(rowah + kb + 16);
      #pragma unroll
      for (int j = 0; j < 4; ++j) { a.u[j] = xa.u[j]; a.u[4 + j] = xb.u[j]; }
    }

    // batch-load all 8 B fragments, then back-to-back WMMAs
    Frag16 b[NT];
    #pragma unroll
    for (int ct = 0; ct < NT; ++ct) {
      const unsigned* bp = &ldsw[(ct * 16 + nlo) * KP2 + kc * 16 + khalf * 4];
      *(uint4*)&b[ct].u[0] = *(const uint4*)bp;
      *(uint4*)&b[ct].u[4] = *(const uint4*)(bp + 8);
    }
    #pragma unroll
    for (int ct = 0; ct < NT; ++ct)
      acc[ct] = __builtin_amdgcn_wmma_f32_16x16x32_bf16(
          false, a.v, false, b[ct].v, (short)0, acc[ct], false, false);
  }

  // ---- epilogue; uniform fast path avoids per-store exec-mask churn ----
  auto epilog = [&](bool guard) {
    #pragma unroll
    for (int ct = 0; ct < NT; ++ct) {
      const int n = ct * 16 + nlo;
      #pragma unroll
      for (int r = 0; r < 8; ++r) {
        const int m = m0 + khalf * 8 + r;
        if (!guard || m < M) {
          float v = acc[ct][r];
          if (addendv) {
            if constexpr (CBF16)
              v += bflo((unsigned)((const unsigned short*)addendv)[(size_t)m * ldc + n]);
            else
              v += ((const float*)addendv)[(size_t)m * ldc + n];
          }
          if (bias) v += bias[n];
          if constexpr (CBF16) {
            const unsigned short pre = f2bf(v);
            if (C2v) ((unsigned short*)C2v)[(size_t)m * ldc + n] = pre;
            ((unsigned short*)Cv)[(size_t)m * ldc + n] =
                (do_relu && v < 0.0f) ? (unsigned short)0 : pre;
          } else {
            if (C2v) ((float*)C2v)[(size_t)m * ldc + n] = v;
            ((float*)Cv)[(size_t)m * ldc + n] = do_relu ? fmaxf(v, 0.0f) : v;
          }
        }
      }
    }
  };
  if (MFULL || m0 + 16 <= M) epilog(false); else epilog(true);
}

// ---- a_message[atom] = sum_{j<4} message[a2b[atom][j]]  (bf16 in/out) ----
__launch_bounds__(256)
__global__ void k_atom_agg(const unsigned short* __restrict__ msg,
                           const int* __restrict__ a2b,
                           unsigned short* __restrict__ amsg, int natoms)
{
  const int t = blockIdx.x * 256 + threadIdx.x;
  const int atom = t >> 4, c = t & 15;    // 16 lanes/atom, 8 bf16 each
  if (atom >= natoms) return;
  const int4 nb = ((const int4*)a2b)[atom];
  const uint4 r0 = ((const uint4*)(msg + (size_t)nb.x * HIDDEN))[c];
  const uint4 r1 = ((const uint4*)(msg + (size_t)nb.y * HIDDEN))[c];
  const uint4 r2 = ((const uint4*)(msg + (size_t)nb.z * HIDDEN))[c];
  const uint4 r3 = ((const uint4*)(msg + (size_t)nb.w * HIDDEN))[c];
  uint4 o;
  o.x = addpk4(r0.x, r1.x, r2.x, r3.x);
  o.y = addpk4(r0.y, r1.y, r2.y, r3.y);
  o.z = addpk4(r0.z, r1.z, r2.z, r3.z);
  o.w = addpk4(r0.w, r1.w, r2.w, r3.w);
  ((uint4*)(amsg + (size_t)atom * HIDDEN))[c] = o;
}

__launch_bounds__(256)
__global__ void k_fill_zero(float* p, unsigned n) {
  const unsigned i = blockIdx.x * 256u + threadIdx.x;
  if (i < n) p[i] = 0.0f;
}

// ---- segment sum pooling via fp32 global atomics ----
__launch_bounds__(256)
__global__ void k_pool(const float* __restrict__ ahid,
                       const int* __restrict__ atom_mol,
                       float* sums, float* counts, int natoms)
{
  const int t = blockIdx.x * 256 + threadIdx.x;
  const int atom = t >> 5, c = t & 31;
  if (atom >= natoms) return;
  const int mol = atom_mol[atom];
  const float4 v = ((const float4*)(ahid + (size_t)atom * HIDDEN))[c];
  float* dst = sums + (size_t)mol * HIDDEN + c * 4;
  atomicAdd(dst + 0, v.x);
  atomicAdd(dst + 1, v.y);
  atomicAdd(dst + 2, v.z);
  atomicAdd(dst + 3, v.w);
  if (c == 0) atomicAdd(counts + mol, 1.0f);
}

// ---- out[mol] = h1[mol,:] . W2 + b2 ----
__launch_bounds__(256)
__global__ void k_ffn2(const float* __restrict__ h1,
                       const float* __restrict__ W2,
                       const float* __restrict__ b2,
                       float* __restrict__ out, int nm)
{
  const int mol = blockIdx.x * 256 + threadIdx.x;
  if (mol >= nm) return;
  const float* hr = h1 + (size_t)mol * FFN_HIDDEN;
  float s = b2[0];
  #pragma unroll 8
  for (int k = 0; k < FFN_HIDDEN; ++k) s += hr[k] * W2[k];
  out[mol] = s;
}

// =====================================================================
extern "C" void kernel_launch(void* const* d_in, const int* in_sizes, int n_in,
                              void* d_out, int out_size, void* d_ws, size_t ws_size,
                              hipStream_t stream) {
  const float* f_atoms  = (const float*)d_in[0];
  const float* f_bonds  = (const float*)d_in[1];
  const int*   a2b      = (const int*)d_in[2];
  const int*   b2a      = (const int*)d_in[3];
  const int*   b2revb   = (const int*)d_in[4];
  const int*   atom_mol = (const int*)d_in[5];
  const float* W_i      = (const float*)d_in[6];
  const float* W_h      = (const float*)d_in[7];
  const float* W_o      = (const float*)d_in[8];
  const float* b_o      = (const float*)d_in[9];
  const float* W1       = (const float*)d_in[10];
  const float* b1       = (const float*)d_in[11];
  const float* W2       = (const float*)d_in[12];
  const float* b2       = (const float*)d_in[13];
  float* out = (float*)d_out;

  // workspace: bf16 message state (WMMA consumes bf16 anyway -> half traffic)
  char* base = (char*)d_ws;
  const size_t msgBytes = (size_t)N_BONDS * HIDDEN * sizeof(unsigned short); // 102.4MB
  unsigned short* inp  = (unsigned short*)base;                  // pre-act bond input
  unsigned short* msgA = (unsigned short*)(base + msgBytes);
  unsigned short* msgB = (unsigned short*)(base + 2 * msgBytes);
  unsigned short* amsg = (unsigned short*)(base + 3 * msgBytes); // [N_ATOMS,128] bf16
  float* ahid   = (float*)base;                    // alias inp (dead after MP loop)
  float* sums   = (float*)(base + msgBytes);       // alias msgA (dead after MP loop)
  float* counts = sums + (size_t)N_MOLS * HIDDEN;
  float* h1     = counts + N_MOLS;                 // [N_MOLS,256]

  const dim3 blk(256);
  const int gb   = (N_BONDS + 127) / 128;          // exact: 400000 % 128 == 0
  const int ga   = (N_ATOMS + 127) / 128;
  const int gm   = (N_MOLS  + 127) / 128;
  const int gagg = (N_ATOMS * 16 + 255) / 256;
  const int gpol = (N_ATOMS * 32 + 255) / 256;

  // 0) inp = f_bonds @ W_i (pre-act, bf16); msgA = relu(inp)
  gemm_wmma_bf16<0, 160, BOND_FDIM, true, true><<<gb, blk, 0, stream>>>(
      f_bonds, nullptr, nullptr, nullptr, W_i, nullptr, nullptr,
      msgA, inp, N_BONDS, BOND_FDIM, 1, HIDDEN, HIDDEN);

  // 1) DEPTH-1 fused message-passing iterations
  const unsigned short* cur = msgA;
  unsigned short* nxt = msgB;
  for (int it = 0; it < DEPTH - 1; ++it) {
    k_atom_agg<<<gagg, blk, 0, stream>>>(cur, a2b, amsg, N_ATOMS);
    // nxt[b] = relu(inp[b] + (amsg[b2a[b]] - cur[b2revb[b]]) @ W_h)
    gemm_wmma_bf16<1, 128, 128, true, true><<<gb, blk, 0, stream>>>(
        amsg, cur, b2a, b2revb, W_h, nullptr, inp, nxt, nullptr,
        N_BONDS, HIDDEN, 1, HIDDEN, HIDDEN);
    unsigned short* t = (unsigned short*)cur; cur = nxt; nxt = t;
  }

  // 2) atom head: relu([f_atoms, amsg] @ W_o + b_o), two accumulating passes
  k_atom_agg<<<gagg, blk, 0, stream>>>(cur, a2b, amsg, N_ATOMS);
  gemm_wmma_bf16<0, 160, ATOM_FDIM, false, false><<<ga, blk, 0, stream>>>(
      f_atoms, nullptr, nullptr, nullptr, W_o, nullptr, nullptr,
      ahid, nullptr, N_ATOMS, ATOM_FDIM, 0, HIDDEN, HIDDEN);
  gemm_wmma_bf16<3, 128, 128, false, false><<<ga, blk, 0, stream>>>(
      amsg, nullptr, nullptr, nullptr, W_o + (size_t)ATOM_FDIM * HIDDEN, b_o,
      ahid, ahid, nullptr, N_ATOMS, HIDDEN, 1, HIDDEN, HIDDEN);

  // 3) per-molecule mean pooling
  const unsigned zn = (unsigned)(N_MOLS * HIDDEN + N_MOLS);
  k_fill_zero<<<(zn + 255) / 256, blk, 0, stream>>>(sums, zn);
  k_pool<<<gpol, blk, 0, stream>>>(ahid, atom_mol, sums, counts, N_ATOMS);

  // 4) FFN1: h1 = relu(mean @ W1 + b1), two 128-column passes
  gemm_wmma_bf16<2, 128, 128, false, false><<<gm, blk, 0, stream>>>(
      sums, counts, nullptr, nullptr, W1, b1, nullptr,
      h1, nullptr, N_MOLS, HIDDEN, 1, FFN_HIDDEN, FFN_HIDDEN);
  gemm_wmma_bf16<2, 128, 128, false, false><<<gm, blk, 0, stream>>>(
      sums, counts, nullptr, nullptr, W1 + 128, b1 + 128, nullptr,
      h1 + 128, nullptr, N_MOLS, HIDDEN, 1, FFN_HIDDEN, FFN_HIDDEN);

  // 5) FFN2
  k_ffn2<<<(N_MOLS + 255) / 256, blk, 0, stream>>>(h1, W2, b2, out, N_MOLS);
}
